// MTREncoder_55903294325218
// MI455X (gfx1250) — matrix operand order repaired and checked by reading, not compiled, not run on hardware
//
#include <hip/hip_runtime.h>

// ---------------------------------------------------------------------------
// MTR encoder for MI455X (gfx1250): bf16 WMMA GEMMs, double-buffered LDS
// staging via CDNA5 async global->LDS copies (ASYNCcnt), fused epilogues.
// ---------------------------------------------------------------------------

typedef __attribute__((ext_vector_type(16))) __bf16 v16bf;
typedef __attribute__((ext_vector_type(8)))  float  v8f;
typedef int vint4 __attribute__((vector_size(16)));

#if defined(__has_builtin)
#if __has_builtin(__builtin_amdgcn_global_load_async_to_lds_b128)
#define HAVE_ASYNC_LDS 1
#endif
#endif
#ifndef HAVE_ASYNC_LDS
#define HAVE_ASYNC_LDS 0
#endif

namespace {

constexpr int S    = 8 * 2048;   // stacked tokens
constexpr int D    = 256;
constexpr int H    = 8;
constexpr int HD   = 32;
constexpr int KNN  = 16;
constexpr int DFF  = 1024;
constexpr int NLAY = 6;

__device__ __forceinline__ unsigned short f2bf(float f) {
  unsigned int u = __float_as_uint(f);
  unsigned int r = (u + 0x7FFFu + ((u >> 16) & 1u)) >> 16;   // round-nearest-even
  return (unsigned short)r;
}
__device__ __forceinline__ float bf2f(unsigned short h) {
  return __uint_as_float(((unsigned int)h) << 16);
}

union Frag { uint4 u4[2]; v16bf v; };

#if HAVE_ASYNC_LDS
__device__ __forceinline__ void async_cp16(const unsigned short* g,
                                           unsigned short* l) {
  // GLOBAL_LOAD_ASYNC_TO_LDS_B128: per-lane 16B memory -> LDS, ASYNCcnt.
  __builtin_amdgcn_global_load_async_to_lds_b128(
      (__attribute__((address_space(1))) vint4*)g,
      (__attribute__((address_space(3))) vint4*)l, 0, 0);
}
__device__ __forceinline__ void wait_async() {
  asm volatile("s_wait_asynccnt 0x0" ::: "memory");
}
#endif

} // namespace

// --- one-time: convert W[Din][Dout] fp32 -> WT[Dout][Din] bf16 --------------
__global__ void wt_convert_kernel(const float* __restrict__ W,
                                  unsigned short* __restrict__ WT,
                                  int Din, int Dout) {
  int idx = blockIdx.x * 256 + threadIdx.x;
  if (idx >= Din * Dout) return;
  int o = idx / Din;
  int i = idx - o * Din;
  WT[idx] = f2bf(W[(size_t)i * Dout + o]);
}

// --- fused sine-PE + bf16 conversion of src ---------------------------------
__global__ void addpe_kernel(const float* __restrict__ src,
                             const float* __restrict__ pos,
                             unsigned short* __restrict__ qkin_bf,
                             unsigned short* __restrict__ src_bf) {
  int idx = blockIdx.x * 256 + threadIdx.x;      // S*D threads
  int s = idx >> 8;
  int d = idx & 255;
  // pe = concat(emb(pos_y), emb(pos_x)); out[2m]=sin, out[2m+1]=cos,
  // dim_t = 10000^(m/64)
  float coord = (d < 128) ? pos[s * 2 + 1] : pos[s * 2 + 0];
  int   dd    = d & 127;
  int   m     = dd >> 1;
  float t     = powf(10000.0f, (float)m * (1.0f / 64.0f));
  float ang   = coord * 6.283185307179586f / t;
  float pe    = (dd & 1) ? cosf(ang) : sinf(ang);
  float sv    = src[idx];
  qkin_bf[idx] = f2bf(sv + pe);
  src_bf[idx]  = f2bf(sv);
}

// --- tiled bf16 WMMA GEMM: C[S x Dout] = A[S x Din] * WT^T + bias ----------
// block = 256 threads (8 waves); block tile 128x128; wave tile 64x32
// (4x2 WMMA 16x16x32); double-buffered LDS, async global->LDS staging.
__global__ __launch_bounds__(256)
void gemm_bf16_wmma(const unsigned short* __restrict__ A,    // [S][Din] bf16
                    const unsigned short* __restrict__ WT,   // [Dout][Din] bf16
                    const float* __restrict__ bias,          // [Dout]
                    float* __restrict__ Cf,                  // fp32 out (or null)
                    unsigned short* __restrict__ Cb,         // bf16 out (or null)
                    int Din, int Dout, float alpha, int relu) {
  __shared__ __align__(16) unsigned short ldsA[2][128 * 32];   // 2 x 8 KB
  __shared__ __align__(16) unsigned short ldsB[2][128 * 32];   // 2 x 8 KB

  const int m0    = blockIdx.y * 128;
  const int n0    = blockIdx.x * 128;
  const int lane  = threadIdx.x & 31;
  const int wave  = threadIdx.x >> 5;
  const int wm    = wave >> 2;        // 0..1 : 64-row strip
  const int wn    = wave & 3;         // 0..3 : 32-col strip
  const int lhalf = lane >> 4;        // 0|1
  const int l15   = lane & 15;
  const int t     = threadIdx.x;

  // staging slots: 512 x 16B per tile, 2 per thread (rows of 32 bf16 = 4 segs)
  const int row0 = t >> 2,            seg0 = t & 3;
  const int row1 = (t + 256) >> 2,    seg1 = t & 3;

  v8f acc[4][2];
#pragma unroll
  for (int i = 0; i < 4; ++i)
#pragma unroll
    for (int j = 0; j < 2; ++j)
#pragma unroll
      for (int v = 0; v < 8; ++v) acc[i][j][v] = 0.0f;

#if HAVE_ASYNC_LDS
  auto stage = [&](int buf, int k0) {
    async_cp16(&A [(size_t)(m0 + row0) * Din + k0 + seg0 * 8], &ldsA[buf][row0 * 32 + seg0 * 8]);
    async_cp16(&A [(size_t)(m0 + row1) * Din + k0 + seg1 * 8], &ldsA[buf][row1 * 32 + seg1 * 8]);
    async_cp16(&WT[(size_t)(n0 + row0) * Din + k0 + seg0 * 8], &ldsB[buf][row0 * 32 + seg0 * 8]);
    async_cp16(&WT[(size_t)(n0 + row1) * Din + k0 + seg1 * 8], &ldsB[buf][row1 * 32 + seg1 * 8]);
  };
  stage(0, 0);
  wait_async();
  __syncthreads();
#else
  uint4 ra0, ra1, rb0, rb1;
  auto loadRegs = [&](int k0) {
    ra0 = *(const uint4*)&A [(size_t)(m0 + row0) * Din + k0 + seg0 * 8];
    ra1 = *(const uint4*)&A [(size_t)(m0 + row1) * Din + k0 + seg1 * 8];
    rb0 = *(const uint4*)&WT[(size_t)(n0 + row0) * Din + k0 + seg0 * 8];
    rb1 = *(const uint4*)&WT[(size_t)(n0 + row1) * Din + k0 + seg1 * 8];
  };
  auto storeRegs = [&](int buf) {
    *(uint4*)&ldsA[buf][row0 * 32 + seg0 * 8] = ra0;
    *(uint4*)&ldsA[buf][row1 * 32 + seg1 * 8] = ra1;
    *(uint4*)&ldsB[buf][row0 * 32 + seg0 * 8] = rb0;
    *(uint4*)&ldsB[buf][row1 * 32 + seg1 * 8] = rb1;
  };
  loadRegs(0);
  storeRegs(0);
  __syncthreads();
#endif

  const int steps = Din >> 5;   // K-steps of 32
  int buf = 0;
  for (int ks = 0; ks < steps; ++ks) {
    const bool has_next = (ks + 1 < steps);
#if HAVE_ASYNC_LDS
    if (has_next) stage(buf ^ 1, (ks + 1) * 32);   // overlaps WMMAs below
#else
    if (has_next) loadRegs((ks + 1) * 32);         // global loads in flight
#endif

    // ---- fragments per ISA 7.12.2 (16-bit A 16x32, B 32x16 layouts) ----
    const unsigned short* la = ldsA[buf];
    const unsigned short* lb = ldsB[buf];
    Frag fa[4], fb[2];
#pragma unroll
    for (int i = 0; i < 4; ++i) {
      int M = wm * 64 + i * 16 + l15;
      fa[i].u4[0] = *(const uint4*)&la[M * 32 + lhalf * 8];        // K 0..15 grp
      fa[i].u4[1] = *(const uint4*)&la[M * 32 + 16 + lhalf * 8];   // K 16..31 grp
    }
#pragma unroll
    for (int j = 0; j < 2; ++j) {
      int Ncol = wn * 32 + j * 16 + l15;
      fb[j].u4[0] = *(const uint4*)&lb[Ncol * 32 + lhalf * 8];
      fb[j].u4[1] = *(const uint4*)&lb[Ncol * 32 + 16 + lhalf * 8];
    }

#pragma unroll
    for (int i = 0; i < 4; ++i)
#pragma unroll
      for (int j = 0; j < 2; ++j)
        acc[i][j] = __builtin_amdgcn_wmma_f32_16x16x32_bf16(
            false, fa[i].v, false, fb[j].v, (short)0, acc[i][j], false, false);

#if HAVE_ASYNC_LDS
    if (has_next) wait_async();
    __syncthreads();
#else
    if (has_next) storeRegs(buf ^ 1);  // writes other buffer: no race w/ readers
    __syncthreads();
#endif
    buf ^= 1;
  }

  // ---- epilogue: D-matrix layout: VGPR v -> M = v + 8*lhalf, N = l15 ----
#pragma unroll
  for (int i = 0; i < 4; ++i) {
#pragma unroll
    for (int j = 0; j < 2; ++j) {
      int rbase = m0 + wm * 64 + i * 16 + lhalf * 8;
      int col   = n0 + wn * 32 + j * 16 + l15;
      float bc  = bias[col];
#pragma unroll
      for (int v = 0; v < 8; ++v) {
        float tv = (acc[i][j][v] + bc) * alpha;
        if (relu) tv = fmaxf(tv, 0.0f);
        size_t off = (size_t)(rbase + v) * Dout + col;
        if (Cf) Cf[off] = tv;
        if (Cb) Cb[off] = f2bf(tv);
      }
    }
  }
}

// --- kNN attention: one thread per (token, head); writes O over Q ----------
__global__ __launch_bounds__(256)
void knn_attn_kernel(const unsigned short* __restrict__ Qb,
                     const unsigned short* __restrict__ Kb,
                     const unsigned short* __restrict__ Vb,
                     const int* __restrict__ idxp,
                     unsigned short* __restrict__ Ob /* == Qb */) {
  int t = blockIdx.x * 256 + threadIdx.x;     // S*H threads
  int s = t >> 3;
  int h = t & 7;
  const int base = s * D + h * HD;

  float q[HD];
#pragma unroll
  for (int d = 0; d < HD; ++d) q[d] = bf2f(Qb[base + d]);

  float lg[KNN];
  int   nb[KNN];
#pragma unroll
  for (int kk = 0; kk < KNN; ++kk) {
    int id = idxp[s * KNN + kk];
    int cid = (id >= 0) ? id : 0;
    nb[kk] = cid;
    float a = 0.0f;
    int kb = cid * D + h * HD;
#pragma unroll
    for (int d = 0; d < HD; ++d) a += q[d] * bf2f(Kb[kb + d]);
    lg[kk] = (id >= 0) ? a : -1e9f;
  }

  float mx = -3.4e38f;
#pragma unroll
  for (int kk = 0; kk < KNN; ++kk) mx = fmaxf(mx, lg[kk]);
  float sum = 0.0f;
  float p[KNN];
#pragma unroll
  for (int kk = 0; kk < KNN; ++kk) { p[kk] = __expf(lg[kk] - mx); sum += p[kk]; }
  float inv = 1.0f / sum;

  float o[HD];
#pragma unroll
  for (int d = 0; d < HD; ++d) o[d] = 0.0f;
#pragma unroll
  for (int kk = 0; kk < KNN; ++kk) {
    float w = p[kk] * inv;
    int vb = nb[kk] * D + h * HD;
#pragma unroll
    for (int d = 0; d < HD; ++d) o[d] += w * bf2f(Vb[vb + d]);
  }
#pragma unroll
  for (int d = 0; d < HD; ++d) Ob[base + d] = f2bf(o[d]);
}

// --- residual add + LayerNorm: one wave32 per token -------------------------
__global__ __launch_bounds__(256)
void residual_ln_kernel(float* __restrict__ src,        // running activation (d_out)
                        const float* __restrict__ res,  // branch output
                        const float* __restrict__ g,
                        const float* __restrict__ b,
                        unsigned short* __restrict__ out_bf /* nullable */) {
  int token = blockIdx.x * 8 + (threadIdx.x >> 5);
  int lane  = threadIdx.x & 31;
  int base  = token * D + lane * 8;

  float v[8];
  float s1 = 0.0f, s2 = 0.0f;
#pragma unroll
  for (int j = 0; j < 8; ++j) {
    float t = src[base + j] + res[base + j];
    v[j] = t; s1 += t; s2 += t * t;
  }
#pragma unroll
  for (int off = 16; off > 0; off >>= 1) {
    s1 += __shfl_xor(s1, off, 32);
    s2 += __shfl_xor(s2, off, 32);
  }
  float mean = s1 * (1.0f / D);
  float var  = s2 * (1.0f / D) - mean * mean;
  float rstd = rsqrtf(var + 1e-5f);
#pragma unroll
  for (int j = 0; j < 8; ++j) {
    int d   = lane * 8 + j;
    float y = (v[j] - mean) * rstd * g[d] + b[d];
    src[base + j] = y;
    if (out_bf) out_bf[base + j] = f2bf(y);
  }
}

// ---------------------------------------------------------------------------
extern "C" void kernel_launch(void* const* d_in, const int* in_sizes, int n_in,
                              void* d_out, int out_size, void* d_ws, size_t ws_size,
                              hipStream_t stream) {
  const float* x    = (const float*)d_in[0];
  const float* pos  = (const float*)d_in[1];
  const int*   idxp = (const int*)d_in[2];
  const float* Wq = (const float*)d_in[3];  const float* bq = (const float*)d_in[4];
  const float* Wk = (const float*)d_in[5];  const float* bk = (const float*)d_in[6];
  const float* Wv = (const float*)d_in[7];  const float* bv = (const float*)d_in[8];
  const float* Wo = (const float*)d_in[9];  const float* bo = (const float*)d_in[10];
  const float* W1 = (const float*)d_in[11]; const float* b1 = (const float*)d_in[12];
  const float* W2 = (const float*)d_in[13]; const float* b2 = (const float*)d_in[14];
  const float* g1 = (const float*)d_in[15]; const float* be1 = (const float*)d_in[16];
  const float* g2 = (const float*)d_in[17]; const float* be2 = (const float*)d_in[18];

  float* out = (float*)d_out;   // running src, [S][D] fp32

  // workspace carve-out
  size_t off = 0;
  auto take = [&](size_t bytes) -> void* {
    void* p = (char*)d_ws + off;
    off += (bytes + 255) & ~(size_t)255;
    return p;
  };
  const size_t PER_LAYER_WT = 4 * 65536 + 2 * 262144;  // 786432 bf16 elems
  unsigned short* WTall = (unsigned short*)take((size_t)NLAY * PER_LAYER_WT * 2);
  unsigned short* qkin  = (unsigned short*)take((size_t)S * D * 2);
  unsigned short* srcb  = (unsigned short*)take((size_t)S * D * 2);
  unsigned short* Qb    = (unsigned short*)take((size_t)S * D * 2);
  unsigned short* Kb    = (unsigned short*)take((size_t)S * D * 2);
  unsigned short* Vb    = (unsigned short*)take((size_t)S * D * 2);
  float*          oproj = (float*)take((size_t)S * D * 4);
  unsigned short* hbuf  = (unsigned short*)take((size_t)S * DFF * 2);

  // init running activation from x
  (void)hipMemcpyAsync(out, x, (size_t)S * D * sizeof(float),
                       hipMemcpyDeviceToDevice, stream);

  // one-time weight convert+transpose to bf16 (deterministic each call)
  for (int l = 0; l < NLAY; ++l) {
    unsigned short* WTl = WTall + (size_t)l * PER_LAYER_WT;
    wt_convert_kernel<<<65536 / 256, 256, 0, stream>>>(Wq + (size_t)l * 65536, WTl + 0,      D, D);
    wt_convert_kernel<<<65536 / 256, 256, 0, stream>>>(Wk + (size_t)l * 65536, WTl + 65536,  D, D);
    wt_convert_kernel<<<65536 / 256, 256, 0, stream>>>(Wv + (size_t)l * 65536, WTl + 131072, D, D);
    wt_convert_kernel<<<65536 / 256, 256, 0, stream>>>(Wo + (size_t)l * 65536, WTl + 196608, D, D);
    wt_convert_kernel<<<262144 / 256, 256, 0, stream>>>(W1 + (size_t)l * 262144, WTl + 262144, D, DFF);
    wt_convert_kernel<<<262144 / 256, 256, 0, stream>>>(W2 + (size_t)l * 262144, WTl + 524288, DFF, D);
  }

  const dim3 blk(256);
  const dim3 gD(D / 128, S / 128);     // Dout=256 GEMMs
  const dim3 gF(DFF / 128, S / 128);   // Dout=1024 GEMM
  const float qscale = 0.17677669529663688f;  // 1/sqrt(hd=32)

  for (int l = 0; l < NLAY; ++l) {
    unsigned short* WTl = WTall + (size_t)l * PER_LAYER_WT;
    unsigned short* WTq = WTl + 0;
    unsigned short* WTk = WTl + 65536;
    unsigned short* WTv = WTl + 131072;
    unsigned short* WTo = WTl + 196608;
    unsigned short* WT1 = WTl + 262144;
    unsigned short* WT2 = WTl + 524288;

    // qk_in = src + PE (on the fly), plus src in bf16
    addpe_kernel<<<(S * D) / 256, blk, 0, stream>>>(out, pos, qkin, srcb);

    // projections (bf16 WMMA); Q fused with 1/sqrt(hd)
    gemm_bf16_wmma<<<gD, blk, 0, stream>>>(qkin, WTq, bq + (size_t)l * D,
                                           nullptr, Qb, D, D, qscale, 0);
    gemm_bf16_wmma<<<gD, blk, 0, stream>>>(qkin, WTk, bk + (size_t)l * D,
                                           nullptr, Kb, D, D, 1.0f, 0);
    gemm_bf16_wmma<<<gD, blk, 0, stream>>>(srcb, WTv, bv + (size_t)l * D,
                                           nullptr, Vb, D, D, 1.0f, 0);

    // kNN attention (O overwrites Q in place)
    knn_attn_kernel<<<(S * H) / 256, blk, 0, stream>>>(Qb, Kb, Vb, idxp, Qb);

    // output projection -> fp32, then residual + LN (emit bf16 for FFN)
    gemm_bf16_wmma<<<gD, blk, 0, stream>>>(Qb, WTo, bo + (size_t)l * D,
                                           oproj, nullptr, D, D, 1.0f, 0);
    residual_ln_kernel<<<S / 8, blk, 0, stream>>>(out, oproj,
                                                  g1 + (size_t)l * D,
                                                  be1 + (size_t)l * D, srcb);

    // FFN: relu(src@W1+b1) -> bf16 hidden; then @W2+b2 -> fp32
    gemm_bf16_wmma<<<gF, blk, 0, stream>>>(srcb, WT1, b1 + (size_t)l * DFF,
                                           nullptr, hbuf, D, DFF, 1.0f, 1);
    gemm_bf16_wmma<<<gD, blk, 0, stream>>>(hbuf, WT2, b2 + (size_t)l * D,
                                           oproj, nullptr, DFF, D, 1.0f, 0);
    residual_ln_kernel<<<S / 8, blk, 0, stream>>>(out, oproj,
                                                  g2 + (size_t)l * D,
                                                  be2 + (size_t)l * D, nullptr);
  }
  (void)in_sizes; (void)n_in; (void)out_size; (void)ws_size;
}
